// SConv_2052994367860
// MI455X (gfx1250) — matrix operand order repaired and checked by reading, not compiled, not run on hardware
//
#include <hip/hip_runtime.h>
#include <hip/hip_bf16.h>

typedef _Float16 v16h __attribute__((ext_vector_type(16)));
typedef _Float16 v8h  __attribute__((ext_vector_type(8)));
typedef _Float16 v2h  __attribute__((ext_vector_type(2)));
typedef float    v8f  __attribute__((ext_vector_type(8)));

union AV { v16h v; v8h q[2]; };
union HV { v8h v; _Float16 h[8]; };

__device__ __forceinline__ v8f wmma_f16(v16h a, v16h b, v8f c) {
    // D(f32,16x16) = A(f16,16x32) * B(f16,32x16) + C
    return __builtin_amdgcn_wmma_f32_16x16x32_f16(
        /*neg_a=*/false, a, /*neg_b=*/false, b,
        /*c_mod=*/(short)0, c, /*reuse_a=*/false, /*reuse_b=*/false);
}

// Native CDNA5 float atomic max, device scope (scatter-max aggregation).
__device__ __forceinline__ void atomic_max_f32(float* addr, float v) {
    asm volatile("global_atomic_max_num_f32 %0, %1, off scope:SCOPE_DEV"
                 :: "v"(addr), "v"(v) : "memory");
}

// ---------------- conversion kernels ----------------

__global__ void cvt_f16_kernel(const float* __restrict__ in,
                               _Float16* __restrict__ out, long n) {
    long stride = (long)gridDim.x * blockDim.x;
    for (long i = (long)blockIdx.x * blockDim.x + threadIdx.x; i < n; i += stride)
        out[i] = (_Float16)in[i];
}

// W[m][i][o] (fp32) -> Wt[m][o][i] (f16); nmat 64x64 matrices.
__global__ void cvt_transpose_kernel(const float* __restrict__ W,
                                     _Float16* __restrict__ Wt, int nmat) {
    long n = (long)nmat << 12;
    long stride = (long)gridDim.x * blockDim.x;
    for (long idx = (long)blockIdx.x * blockDim.x + threadIdx.x; idx < n; idx += stride) {
        long m = idx >> 12;
        int  i = (int)((idx >> 6) & 63);
        int  o = (int)(idx & 63);
        Wt[(m << 12) + ((long)o << 6) + i] = (_Float16)W[idx];
    }
}

__global__ void fill_f32_kernel(float* __restrict__ p, float v, long n) {
    long stride = (long)gridDim.x * blockDim.x;
    for (long i = (long)blockIdx.x * blockDim.x + threadIdx.x; i < n; i += stride)
        p[i] = v;
}

// ---------------- xk = A @ W[k]  (25 WMMA GEMMs per node tile) ----------------
// A:  [N][64] f16 node features
// Wt: [25][64][64] f16, out-major (Wt[k][o][i])
// xk: [N][25][64] f16  (node-major so edge gathers are one 128B line; 160MB -> fits L2)
__global__ void xk_gemm_kernel(const _Float16* __restrict__ A,
                               const _Float16* __restrict__ Wt,
                               _Float16* __restrict__ xk, int N) {
    const int tile = blockIdx.x;            // 16 nodes per tile
    const int wave = threadIdx.x >> 5;      // 4 waves -> 4 output col tiles of 16
    const int lane = threadIdx.x & 31;
    const int lo16 = lane & 15;
    const int hi   = lane >> 4;

    int row = tile * 16 + lo16;
    if (row >= N) row = N - 1;

    const int koff  = hi ? 16 : 0;          // A-matrix K offset
    const int koffb = hi ? 16 : 0;          // B-matrix K offset

    // A 16x32 f16 layout: lane holds row M=lane%16, K = koff..koff+15 contiguous
    const _Float16* arow = A + (size_t)row * 64;
    AV a0, a1;
    a0.q[0] = *(const v8h*)(arow + koff);
    a0.q[1] = *(const v8h*)(arow + 8 + koff);
    a1.q[0] = *(const v8h*)(arow + 32 + koff);
    a1.q[1] = *(const v8h*)(arow + 40 + koff);

    const int  m0       = hi * 8;
    const int  nodeBase = tile * 16 + m0;
    const int  ch       = wave * 16 + lo16;
    const bool fullTile = (tile * 16 + 16) <= N;   // wave-uniform

    // B 32x16 layout: lane holds column N=lane%16, K = koffb..koffb+15 contiguous
    const _Float16* bbase = Wt + (size_t)ch * 64;  // k=0 column for this lane
    _Float16*       outp  = xk + (size_t)nodeBase * 25 * 64 + ch;

    if (fullTile) {
        // software-pipelined: prefetch next k's B while current WMMAs run
        v16h b0 = *(const v16h*)(bbase + koffb);
        v16h b1 = *(const v16h*)(bbase + 32 + koffb);
        for (int k = 0; k < 25; ++k) {
            const int kn = (k < 24) ? (k + 1) : 24;       // branch-free clamp
            const _Float16* nb = bbase + (size_t)kn * 4096;
            v16h nb0 = *(const v16h*)(nb + koffb);
            v16h nb1 = *(const v16h*)(nb + 32 + koffb);

            v8f c = {0.f, 0.f, 0.f, 0.f, 0.f, 0.f, 0.f, 0.f};
            c = wmma_f16(a0.v, b0, c);
            c = wmma_f16(a1.v, b1, c);

            // convert all 8 first into packed halves (4 VGPRs) so the 8 stores
            // use distinct source regs -> one store clause, no xcnt interlocks
            HV hv;
            #pragma unroll
            for (int r = 0; r < 8; ++r)
                hv.h[r] = (_Float16)c[r];

            // D f32 16x16: VGPR r -> M = m0 + r, N = lane%16
            _Float16* op = outp + (size_t)k * 64;
            #pragma unroll
            for (int r = 0; r < 8; ++r)
                op[(size_t)r * 25 * 64] = hv.h[r];

            b0 = nb0;
            b1 = nb1;
        }
    } else {
        for (int k = 0; k < 25; ++k) {
            const _Float16* bcol = bbase + (size_t)k * 4096;
            v16h b0 = *(const v16h*)(bcol + koffb);
            v16h b1 = *(const v16h*)(bcol + 32 + koffb);

            v8f c = {0.f, 0.f, 0.f, 0.f, 0.f, 0.f, 0.f, 0.f};
            c = wmma_f16(a0.v, b0, c);
            c = wmma_f16(a1.v, b1, c);

            _Float16* op = outp + (size_t)k * 64;
            #pragma unroll
            for (int r = 0; r < 8; ++r)
                if (nodeBase + r < N)
                    op[(size_t)r * 25 * 64] = (_Float16)c[r];
        }
    }
}

// ---------------- per-edge B-spline message + scatter-max ----------------
// one wave per edge; lane covers channels {2*lane, 2*lane+1}
__global__ void edge_scatter_kernel(const _Float16* __restrict__ xk,
                                    const int* __restrict__ src,
                                    const int* __restrict__ dst,
                                    const float* __restrict__ attr,
                                    float* __restrict__ agg, int E) {
    const int e    = blockIdx.x * (blockDim.x >> 5) + (threadIdx.x >> 5);
    const int lane = threadIdx.x & 31;
    if (e >= E) return;

    const int s = src[e];
    const int d = dst[e];
    const float u0 = attr[2 * e + 0] * 4.0f;   // K-1 = 4
    const float u1 = attr[2 * e + 1] * 4.0f;
    const float fb0 = floorf(u0), fb1 = floorf(u1);
    const float f0 = u0 - fb0,    f1 = u1 - fb1;
    const int   i0 = (int)fb0,    i1 = (int)fb1;

    float acc0 = 0.f, acc1 = 0.f;
    #pragma unroll
    for (int s1 = 0; s1 < 2; ++s1) {
        #pragma unroll
        for (int s0 = 0; s0 < 2; ++s0) {
            const float w = (s0 ? f0 : 1.0f - f0) * (s1 ? f1 : 1.0f - f1);
            int k0 = min(max(i0 + s0, 0), 4);
            int k1 = min(max(i1 + s1, 0), 4);
            int k  = k0 + 5 * k1;
            const v2h hv = *(const v2h*)(xk + ((size_t)s * 25 + k) * 64 + 2 * lane);
            acc0 += w * (float)hv.x;
            acc1 += w * (float)hv.y;
        }
    }
    float* ap = agg + (size_t)d * 64 + 2 * lane;
    atomic_max_f32(ap + 0, acc0);
    atomic_max_f32(ap + 1, acc1);
}

// ---------------- root GEMM + finalize (agg fixup, bias, optional ReLU) ----------------
template <bool RELU, typename OUT_T>
__global__ void root_finalize_kernel(const _Float16* __restrict__ A,
                                     const _Float16* __restrict__ rootT,  // [64][64] f16, out-major
                                     const float* __restrict__ bias,
                                     const float* __restrict__ agg,
                                     OUT_T* __restrict__ out, int N) {
    const int tile = blockIdx.x;
    const int wave = threadIdx.x >> 5;
    const int lane = threadIdx.x & 31;
    const int lo16 = lane & 15;
    const int hi   = lane >> 4;

    int row = tile * 16 + lo16;
    if (row >= N) row = N - 1;

    const int koff  = hi ? 16 : 0;
    const int koffb = hi ? 16 : 0;

    const _Float16* arow = A + (size_t)row * 64;
    AV a0, a1;
    a0.q[0] = *(const v8h*)(arow + koff);
    a0.q[1] = *(const v8h*)(arow + 8 + koff);
    a1.q[0] = *(const v8h*)(arow + 32 + koff);
    a1.q[1] = *(const v8h*)(arow + 40 + koff);

    const int ch = wave * 16 + lo16;
    const _Float16* bcol = rootT + (size_t)ch * 64;
    v16h b0 = *(const v16h*)(bcol + koffb);
    v16h b1 = *(const v16h*)(bcol + 32 + koffb);

    v8f c = {0.f, 0.f, 0.f, 0.f, 0.f, 0.f, 0.f, 0.f};
    c = wmma_f16(a0.v, b0, c);
    c = wmma_f16(a1.v, b1, c);

    const int  nodeBase = tile * 16 + hi * 8;
    const bool fullTile = (tile * 16 + 16) <= N;
    const float bv = bias[ch];

    if (fullTile) {
        // compute all 8 values first (distinct regs), then store burst
        OUT_T vals[8];
        #pragma unroll
        for (int r = 0; r < 8; ++r) {
            float av = agg[(size_t)(nodeBase + r) * 64 + ch];
            av = (av > -3.0e38f) ? av : 0.0f;      // -inf (no in-edges) -> 0
            float val = av + c[r] + bv;
            if (RELU) val = fmaxf(val, 0.0f);
            vals[r] = (OUT_T)val;
        }
        #pragma unroll
        for (int r = 0; r < 8; ++r)
            out[(size_t)(nodeBase + r) * 64 + ch] = vals[r];
    } else {
        #pragma unroll
        for (int r = 0; r < 8; ++r) {
            int node = nodeBase + r;
            if (node < N) {
                const size_t idx = (size_t)node * 64 + ch;
                float av = agg[idx];
                av = (av > -3.0e38f) ? av : 0.0f;
                float val = av + c[r] + bv;
                if (RELU) val = fmaxf(val, 0.0f);
                out[idx] = (OUT_T)val;
            }
        }
    }
}

// ---------------- host launch ----------------

static inline size_t align_up(size_t x, size_t a) { return (x + a - 1) & ~(a - 1); }

extern "C" void kernel_launch(void* const* d_in, const int* in_sizes, int n_in,
                              void* d_out, int out_size, void* d_ws, size_t ws_size,
                              hipStream_t stream) {
    const float* x    = (const float*)d_in[0];
    const int*   ei   = (const int*)  d_in[1];
    const float* attr = (const float*)d_in[2];
    const float* W1   = (const float*)d_in[3];
    const float* r1   = (const float*)d_in[4];
    const float* b1   = (const float*)d_in[5];
    const float* W2   = (const float*)d_in[6];
    const float* r2   = (const float*)d_in[7];
    const float* b2   = (const float*)d_in[8];

    const int N = in_sizes[0] / 64;
    const int E = in_sizes[1] / 2;
    const int* src = ei;
    const int* dst = ei + E;

    // workspace layout
    char* p = (char*)d_ws;
    const size_t szNodeH = align_up((size_t)N * 64 * sizeof(_Float16), 256);
    const size_t szW     = align_up((size_t)25 * 64 * 64 * sizeof(_Float16), 256);
    const size_t szRoot  = align_up((size_t)64 * 64 * sizeof(_Float16), 256);
    const size_t szAgg   = align_up((size_t)N * 64 * sizeof(float), 256);

    _Float16* xh  = (_Float16*)p; p += szNodeH;
    _Float16* hh  = (_Float16*)p; p += szNodeH;
    _Float16* Wt1 = (_Float16*)p; p += szW;
    _Float16* Wt2 = (_Float16*)p; p += szW;
    _Float16* rt1 = (_Float16*)p; p += szRoot;
    _Float16* rt2 = (_Float16*)p; p += szRoot;
    float*    agg = (float*)p;    p += szAgg;
    _Float16* xk  = (_Float16*)p; // [N][25][64] f16  (~160 MB)

    const long nNode = (long)N * 64;
    const int  tiles = (N + 15) / 16;
    const int  edgeBlocks = (E + 7) / 8;           // 8 waves (edges) per 256-thread block

    // --- conversions ---
    cvt_f16_kernel<<<(int)((nNode + 255) / 256), 256, 0, stream>>>(x, xh, nNode);
    cvt_transpose_kernel<<<512, 256, 0, stream>>>(W1, Wt1, 25);
    cvt_transpose_kernel<<<512, 256, 0, stream>>>(W2, Wt2, 25);
    cvt_transpose_kernel<<<32, 256, 0, stream>>>(r1, rt1, 1);
    cvt_transpose_kernel<<<32, 256, 0, stream>>>(r2, rt2, 1);

    // ---------- layer 1 ----------
    fill_f32_kernel<<<(int)((nNode + 255) / 256), 256, 0, stream>>>(agg, -INFINITY, nNode);
    xk_gemm_kernel<<<tiles, 128, 0, stream>>>(xh, Wt1, xk, N);
    edge_scatter_kernel<<<edgeBlocks, 256, 0, stream>>>(xk, src, dst, attr, agg, E);
    root_finalize_kernel<true, _Float16><<<tiles, 128, 0, stream>>>(xh, rt1, b1, agg, hh, N);

    // ---------- layer 2 ----------
    fill_f32_kernel<<<(int)((nNode + 255) / 256), 256, 0, stream>>>(agg, -INFINITY, nNode);
    xk_gemm_kernel<<<tiles, 128, 0, stream>>>(hh, Wt2, xk, N);
    edge_scatter_kernel<<<edgeBlocks, 256, 0, stream>>>(xk, src, dst, attr, agg, E);
    root_finalize_kernel<false, float><<<tiles, 128, 0, stream>>>(hh, rt2, b2, agg, (float*)d_out, N);
}
